// DeepPotEmbedding_21423296873076
// MI455X (gfx1250) — matrix operand order repaired and checked by reading, not compiled, not run on hardware
//
#include <hip/hip_runtime.h>
#include <hip/hip_bf16.h>

typedef __attribute__((ext_vector_type(16))) _Float16 v16h;
typedef __attribute__((ext_vector_type(8)))  _Float16 v8h;
typedef __attribute__((ext_vector_type(8)))  float    v8f;
typedef __attribute__((ext_vector_type(2)))  float    v2f;

#define NNODE  50000
#define NEDGE  1600000
#define ZMAX   16
#define DIM    64
#define SUBDIM 8
#define HIDDEN 64

// ws layout (in halves): B-fragments for each weight, each fragment = 512 halves
// (lane-major: frag[lane*16 + h]).  W1: 4 frags @0, W2: 8 @2048, W3: 8 @6144, W4: 8 @10240
#define WOFF1 0
#define WOFF2 2048
#define WOFF3 6144
#define WOFF4 10240
#define WTOTAL 14336          // halves = 28 KB

// ---------------------------------------------------------------------------
// Pack fp32 weights (KxN row-major) into f16 WMMA B-fragments.
// B layout (16x16x32 f16): lane l holds column N = l&15; half h holds
// K = (h<8 ? 0 : 16) + (l>=16 ? 8 : 0) + (h&7), per K-tile of 32.
// ---------------------------------------------------------------------------
__global__ void prep_weights(const float* __restrict__ W1, const float* __restrict__ W2,
                             const float* __restrict__ W3, const float* __restrict__ W4,
                             _Float16* __restrict__ ws) {
  int idx = blockIdx.x * blockDim.x + threadIdx.x;
  if (idx >= WTOTAL) return;
  const float* W; int Krows, base, rem;
  if (idx < WOFF2)      { W = W1; Krows = 1 + ZMAX; base = WOFF1; rem = idx - WOFF1; }
  else if (idx < WOFF3) { W = W2; Krows = HIDDEN;   base = WOFF2; rem = idx - WOFF2; }
  else if (idx < WOFF4) { W = W3; Krows = HIDDEN;   base = WOFF3; rem = idx - WOFF3; }
  else                  { W = W4; Krows = HIDDEN;   base = WOFF4; rem = idx - WOFF4; }
  int f    = rem >> 9;          // fragment index: f = kt*4 + nt
  int e    = rem & 511;
  int lane = e >> 4;
  int h    = e & 15;
  int kt   = f >> 2;
  int nt   = f & 3;
  int K = kt * 32 + ((h < 8) ? 0 : 16) + ((lane >= 16) ? 8 : 0) + (h & 7);
  int N = nt * 16 + (lane & 15);
  float v = (K < Krows) ? W[K * 64 + N] : 0.0f;
  ws[base + rem] = (_Float16)v;
}

// Zero the GRi accumulator living in the low 256 floats of each node's
// 512-float output slot.
__global__ void zero_gri(float* __restrict__ out) {
  out[(long)blockIdx.x * 512 + threadIdx.x] = 0.0f;   // 256 threads/block
}

// Fast SiLU: v_rcp_f32 instead of the IEEE v_div_scale/div_fmas sequence.
__device__ __forceinline__ float silu(float x) {
  return x * __builtin_amdgcn_rcpf(1.0f + __expf(-x));
}

// One hidden layer: C(16x64) = A(16xK) * W(Kx64) + bias, optional SiLU.
// B fragments are fed from LDS (ds_load_b128 pairs).
template <int NK>
__device__ __forceinline__ void wmma_layer(const _Float16* frag,
                                           const float* __restrict__ bias,
                                           const v16h* A, int lane, bool act,
                                           v8f out[4]) {
  int lrow = lane & 15;
#pragma unroll
  for (int nt = 0; nt < 4; ++nt) {
    float bv = bias[nt * 16 + lrow];
    v8f c = {bv, bv, bv, bv, bv, bv, bv, bv};
#pragma unroll
    for (int kt = 0; kt < NK; ++kt) {
      v16h b = *(const v16h*)(frag + (kt * 4 + nt) * 512 + lane * 16);
      c = __builtin_amdgcn_wmma_f32_16x16x32_f16(false, A[kt], false, b,
                                                 (short)0, c, false, false);
    }
    if (act) {
#pragma unroll
      for (int r = 0; r < 8; ++r) c[r] = silu(c[r]);
    }
    out[nt] = c;
  }
}

// C layout (f32 16x16): elem r -> row M = r + (lane>=16)*8, col N = lane&15.
// Bounce the 16x64 activation tile through per-wave LDS (row-major halves)
// and reload as two 16x32 A fragments:
// A half h -> K = kt*32 + (h<8?0:16) + (lane>=16)*8 + (h&7)  (two b128 loads).
__device__ __forceinline__ void relayout(const v8f c[4], _Float16* L,
                                         int lane, v16h A[2]) {
  int lrow = lane & 15;
  int hi   = lane >> 4;
#pragma unroll
  for (int nt = 0; nt < 4; ++nt)
#pragma unroll
    for (int r = 0; r < 8; ++r)
      L[(r + hi * 8) * 64 + nt * 16 + lrow] = (_Float16)c[nt][r];
#pragma unroll
  for (int kt = 0; kt < 2; ++kt) {
    v8h lo = *(const v8h*)&L[lrow * 64 + kt * 32 + hi * 8];
    v8h hh = *(const v8h*)&L[lrow * 64 + kt * 32 + 16 + hi * 8];
    A[kt] = __builtin_shufflevector(lo, hh, 0, 1, 2, 3, 4, 5, 6, 7,
                                    8, 9, 10, 11, 12, 13, 14, 15);
  }
}

// ---------------------------------------------------------------------------
// Edge MLP + scatter.  One wave32 = one 16-edge tile; block = 8 waves.
// Weights are DMA'd once per block into LDS with the gfx1250 async
// global->LDS path, then all B operands come from ds_load_b128.
// ---------------------------------------------------------------------------
__global__ void __launch_bounds__(256, 2)
mlp_scatter(const int* __restrict__ species, const int* __restrict__ edge_src,
            const int* __restrict__ edge_dst, const float* __restrict__ dist,
            const float* __restrict__ swt, const float* __restrict__ vec,
            const float* __restrict__ b1, const float* __restrict__ b2,
            const float* __restrict__ b3, const float* __restrict__ b4,
            const _Float16* __restrict__ wf, float* __restrict__ gri /* node*512 */) {
  __shared__ _Float16 wlds[WTOTAL];      // 28 KB: all weight fragments
  __shared__ _Float16 stage[8][16 * 64]; // 16 KB: per-wave relayout staging

  // --- Async-copy the 28 KB weight fragment blob into LDS (1792 x b128). ---
  {
    unsigned long long gbase = (unsigned long long)(size_t)wf;
    for (int i = threadIdx.x; i < WTOTAL / 8; i += 256) {
      unsigned ldsaddr = (unsigned)(size_t)(&wlds[i * 8]);  // low 32 bits = LDS offset
      unsigned goff    = (unsigned)(i * 16);                // byte offset into blob
      asm volatile("global_load_async_to_lds_b128 %0, %1, %2"
                   :: "v"(ldsaddr), "v"(goff), "s"(gbase) : "memory");
    }
    asm volatile("s_wait_asynccnt 0" ::: "memory");
  }
  __syncthreads();

  int wave = threadIdx.x >> 5;
  int lane = threadIdx.x & 31;
  int lrow = lane & 15;
  int hi   = lane >> 4;
  long tile  = (long)blockIdx.x * 8 + wave;
  long ebase = tile * 16;

  // Prefetch the edge-geometry stream consumed in the scatter phase.
  __builtin_prefetch(&vec[ebase * 3], 0, 0);
  __builtin_prefetch(&edge_src[ebase], 0, 0);

  // Build layer-1 input fragment A (16 edges x 32, K = [sij | onehot(spec) | 0pad]).
  // A layout: row M = lane&15; half h -> K = (h<8?0:16) + (lane>=16)*8 + (h&7).
  long m0 = ebase + lrow;
  float sij0 = swt[m0] * __builtin_amdgcn_rcpf(dist[m0]);
  int spec0  = species[edge_dst[m0]];
  v16h A1;
#pragma unroll
  for (int h = 0; h < 16; ++h) {
    int K = ((h < 8) ? 0 : 16) + hi * 8 + (h & 7);
    float v = (K == 0) ? sij0
            : ((K <= ZMAX) ? ((spec0 == (K - 1)) ? 1.0f : 0.0f) : 0.0f);
    A1[h] = (_Float16)v;
  }

  v8f  h4[4];
  v16h A[2];
  wmma_layer<1>(wlds + WOFF1, b1, &A1, lane, true, h4);   //  4 WMMA
  relayout(h4, stage[wave], lane, A);
  wmma_layer<2>(wlds + WOFF2, b2, A, lane, true, h4);     //  8 WMMA
  relayout(h4, stage[wave], lane, A);
  wmma_layer<2>(wlds + WOFF3, b3, A, lane, true, h4);     //  8 WMMA
  relayout(h4, stage[wave], lane, A);
  wmma_layer<2>(wlds + WOFF4, b4, A, lane, false, h4);    //  8 WMMA -> Gij

  // Scatter: GRi[src][a][ch] += Gij[edge][ch] * Rij[edge][a].
  // C elem r of tile nt -> edge M = r + hi*8, channel ch = nt*16 + lrow.
#pragma unroll
  for (int r = 0; r < 8; ++r) {
    long e   = ebase + r + hi * 8;
    float inv = __builtin_amdgcn_rcpf(dist[e]);
    float sij = swt[e] * inv;
    float R0 = sij;
    float R1 = sij * vec[e * 3 + 0] * inv;
    float R2 = sij * vec[e * 3 + 1] * inv;
    float R3 = sij * vec[e * 3 + 2] * inv;
    float* gb = gri + (long)edge_src[e] * 512;
#pragma unroll
    for (int nt = 0; nt < 4; ++nt) {
      float g = h4[nt][r];
      int ch  = nt * 16 + lrow;
      atomicAdd(gb + 0 * 64 + ch, g * R0);
      atomicAdd(gb + 1 * 64 + ch, g * R1);
      atomicAdd(gb + 2 * 64 + ch, g * R2);
      atomicAdd(gb + 3 * 64 + ch, g * R3);
    }
  }
}

// ---------------------------------------------------------------------------
// Final per-node einsum: emb(64x8) = GRi^T(64x4) @ GRisub(4x8) via
// V_WMMA_F32_16X16X4_F32 (N padded to 16).  One wave32 = one node.
// A f32 16x4: lane -> M = lane&15, elem e -> K = (lane>=16)*2 + e.
// B f32 4x16: lane -> N = lane&15, elem e -> K = (lane>=16)*2 + e.
// ---------------------------------------------------------------------------
__global__ void __launch_bounds__(256)
embed(const float* __restrict__ griAll, float* __restrict__ out) {
  int wave = threadIdx.x >> 5;
  int lane = threadIdx.x & 31;
  int lrow = lane & 15;
  int hi   = lane >> 4;
  long node = (long)blockIdx.x * 8 + wave;
  const float* G = griAll + node * 512;    // G[a*64 + d], a<4, d<64 (low 256)

  v2f Bf;
  Bf[0] = (lrow < SUBDIM) ? G[(hi * 2 + 0) * 64 + lrow] : 0.0f;
  Bf[1] = (lrow < SUBDIM) ? G[(hi * 2 + 1) * 64 + lrow] : 0.0f;

  v8f res[4];
#pragma unroll
  for (int mt = 0; mt < 4; ++mt) {
    int d = mt * 16 + lrow;
    v2f Af;
    Af[0] = G[(hi * 2 + 0) * 64 + d];
    Af[1] = G[(hi * 2 + 1) * 64 + d];
    v8f cz = {0.f, 0.f, 0.f, 0.f, 0.f, 0.f, 0.f, 0.f};
    res[mt] = __builtin_amdgcn_wmma_f32_16x16x4_f32(false, Af, false, Bf,
                                                    (short)0, cz, false, false);
  }

  // All loads above are consumed by the WMMAs before these stores issue,
  // so overwriting the GRi region of this node is safe (same wave owns it).
  float* O = out + node * 512;
  if (lrow < SUBDIM) {
#pragma unroll
    for (int mt = 0; mt < 4; ++mt)
#pragma unroll
      for (int r = 0; r < 8; ++r) {
        int d = mt * 16 + r + hi * 8;
        O[d * SUBDIM + lrow] = res[mt][r];
      }
  }
}

extern "C" void kernel_launch(void* const* d_in, const int* in_sizes, int n_in,
                              void* d_out, int out_size, void* d_ws, size_t ws_size,
                              hipStream_t stream) {
  const int*   species  = (const int*)d_in[0];
  const int*   edge_src = (const int*)d_in[1];
  const int*   edge_dst = (const int*)d_in[2];
  const float* dist     = (const float*)d_in[3];
  const float* swt      = (const float*)d_in[4];
  const float* vec      = (const float*)d_in[5];
  const float* W1 = (const float*)d_in[6];
  const float* b1 = (const float*)d_in[7];
  const float* W2 = (const float*)d_in[8];
  const float* b2 = (const float*)d_in[9];
  const float* W3 = (const float*)d_in[10];
  const float* b3 = (const float*)d_in[11];
  const float* W4 = (const float*)d_in[12];
  const float* b4 = (const float*)d_in[13];
  float*     out   = (float*)d_out;
  _Float16*  wfrag = (_Float16*)d_ws;

  prep_weights<<<(WTOTAL + 255) / 256, 256, 0, stream>>>(W1, W2, W3, W4, wfrag);
  zero_gri<<<NNODE, 256, 0, stream>>>(out);
  mlp_scatter<<<NEDGE / (16 * 8), 256, 0, stream>>>(
      species, edge_src, edge_dst, dist, swt, vec,
      b1, b2, b3, b4, wfrag, out);
  embed<<<NNODE / 8, 256, 0, stream>>>(out, out);
}
